// MultiBoxloss_68221260529839
// MI455X (gfx1250) — compile-verified, matrix operand order checked by reading, not run on hardware
//
#include <hip/hip_runtime.h>
#include <math.h>

typedef __attribute__((ext_vector_type(16))) _Float16 v16h;
typedef __attribute__((ext_vector_type(8)))  float    v8f;

#define BB 32
#define PP 16384
#define OO 16
#define CC 81
#define GY3 128          // grid.y for loss kernel: PP / (8 waves * 16 rows)
#define MINE_T 1024

typedef unsigned long long u64;
typedef unsigned int u32;

__device__ __forceinline__ u64 pack_ov(float ov, int p) {
    // ov >= 0 always -> float bits are monotone; smaller p wins ties
    return ((u64)__float_as_uint(ov) << 32) | (u64)(0xFFFFFFFFu - (u32)p);
}

// ---------------- init ----------------
__global__ void k_init(u64* bestPrior, int* numpos) {
    int t = threadIdx.x;
    if (t < BB * OO) bestPrior[t] = 0ull;
    if (t < BB) numpos[t] = 0;
}

// ---------------- match: IoU, per-prior best truth, per-truth best prior ----------------
__global__ void k_match(const float* __restrict__ priors,
                        const float* __restrict__ targets,
                        float* __restrict__ bto, int* __restrict__ bti,
                        u64* __restrict__ bestPrior) {
    __shared__ float st[OO][5];          // x1,y1,x2,y2,area (point_form of truths)
    __shared__ u64 sbp[OO];
    const int b = blockIdx.x;
    const int tid = threadIdx.x;
    if (tid < OO) {
        const float* tg = targets + ((size_t)b * OO + tid) * 5;
        float cx = tg[1], cy = tg[2], w = tg[3], h = tg[4];
        st[tid][0] = cx - 0.5f * w; st[tid][1] = cy - 0.5f * h;
        st[tid][2] = cx + 0.5f * w; st[tid][3] = cy + 0.5f * h;
        st[tid][4] = w * h;
        sbp[tid] = 0ull;
    }
    __syncthreads();

    const int p = blockIdx.y * blockDim.x + tid;
    const float* pr = priors + (size_t)p * 4;
    float pcx = pr[0], pcy = pr[1], pw = pr[2], ph = pr[3];
    float px1 = pcx - 0.5f * pw, py1 = pcy - 0.5f * ph;
    float px2 = pcx + 0.5f * pw, py2 = pcy + 0.5f * ph;
    float pa = pw * ph;

    float bov = -1.0f; int bidx = 0;
    #pragma unroll
    for (int t = 0; t < OO; ++t) {
        float iw = fminf(px2, st[t][2]) - fmaxf(px1, st[t][0]);
        float ih = fminf(py2, st[t][3]) - fmaxf(py1, st[t][1]);
        iw = fmaxf(iw, 0.0f); ih = fmaxf(ih, 0.0f);
        float inter = iw * ih;
        float iou = inter / (pa + st[t][4] - inter);
        if (iou > bov) { bov = iou; bidx = t; }     // first-max tie rule
        atomicMax(&sbp[t], pack_ov(iou, p));
    }
    bto[(size_t)b * PP + p] = bov;
    bti[(size_t)b * PP + p] = bidx;
    __syncthreads();
    if (tid < OO) atomicMax(&bestPrior[b * OO + tid], sbp[tid]);
}

// ---------------- force-match scatter (last-wins, like JAX scatter) ----------------
__global__ void k_force(const u64* __restrict__ bestPrior,
                        float* __restrict__ bto, int* __restrict__ bti) {
    const int b = blockIdx.x;
    for (int t = 0; t < OO; ++t) {
        u64 pk = bestPrior[b * OO + t];
        int bp = (int)(0xFFFFFFFFu - (u32)(pk & 0xFFFFFFFFull));
        bto[(size_t)b * PP + bp] = 2.0f;
        bti[(size_t)b * PP + bp] = t;
    }
}

// ---------------- fused pass: WMMA sum-exp, CE, smooth-L1, rank values ----------------
__global__ void k_loss(const float* __restrict__ conf,
                       const float* __restrict__ loc,
                       const float* __restrict__ priors,
                       const float* __restrict__ targets,
                       const float* __restrict__ bto,
                       const int* __restrict__ bti,
                       float* __restrict__ rank,
                       float* __restrict__ partialL,
                       float* __restrict__ partialC,
                       int* __restrict__ numpos) {
    __shared__ float rL[256], rC[256];
    __shared__ int   rN[256];

    const int tid  = threadIdx.x;
    const int wave = tid >> 5;
    const int lane = tid & 31;
    const int row  = lane & 15;
    const int half = lane >> 4;            // lanes 0-15 / 16-31 both own row = lane%16
    const int b    = blockIdx.x;
    const int p    = (blockIdx.y * 8 + wave) * 16 + row;

    const size_t bp = (size_t)b * PP + p;
    const float* rowp = conf + bp * CC;
    __builtin_prefetch(rowp, 0, 0);        // global_prefetch_b8

    // row max over 81 classes, split across the two lanes owning this row
    float m = -3.0e38f;
    {
        int c0 = half ? 41 : 0, c1 = half ? CC : 41;
        for (int c = c0; c < c1; ++c) m = fmaxf(m, rowp[c]);
        m = fmaxf(m, __shfl_xor(m, 16, 32));
    }

    // WMMA row-sum of exp(x - m): A = f16 exp tiles, B = ones, D[m][*] = sum_k A[m][k]
    v16h bones;
    #pragma unroll
    for (int i = 0; i < 16; ++i) bones[i] = (_Float16)1.0f;
    v8f acc = {};
    const int h = half * 8;
    #pragma unroll
    for (int cb = 0; cb < 96; cb += 32) {
        v16h a;
        #pragma unroll
        for (int i = 0; i < 8; ++i) {
            int c0 = cb + h + i;
            int c1 = cb + h + 16 + i;
            float e0 = (c0 < CC) ? __expf(rowp[c0] - m) : 0.0f;
            float e1 = (c1 < CC) ? __expf(rowp[c1] - m) : 0.0f;
            a[i]     = (_Float16)e0;
            a[i + 8] = (_Float16)e1;
        }
        acc = __builtin_amdgcn_wmma_f32_16x16x32_f16(false, a, false, bones,
                                                     (short)0, acc, false, false);
    }
    // extract this row's sum: row r lives in VGPR (r&7), low lanes if r<8 else high lanes
    float sume;
    {
        float t0 = acc[0], t1 = acc[1], t2 = acc[2], t3 = acc[3];
        float t4 = acc[4], t5 = acc[5], t6 = acc[6], t7 = acc[7];
        int vg = row & 7;
        float x = t0;
        x = (vg == 1) ? t1 : x; x = (vg == 2) ? t2 : x; x = (vg == 3) ? t3 : x;
        x = (vg == 4) ? t4 : x; x = (vg == 5) ? t5 : x; x = (vg == 6) ? t6 : x;
        x = (vg == 7) ? t7 : x;
        float y = __shfl_xor(x, 16, 32);
        bool rowLow  = (row < 8);
        bool laneLow = (lane < 16);
        sume = (rowLow == laneLow) ? x : y;
    }
    float lse = m + __logf(sume);

    float ll = 0.0f, pce = 0.0f; int np = 0;
    float rv = 0.0f; bool doWrite = (half == 0);
    if (doWrite) {
        float ov = bto[bp];
        int   ti = bti[bp];
        const float* tg = targets + ((size_t)b * OO + ti) * 5;
        int cls = (ov < 0.5f) ? 0 : (int)tg[0];
        float ce = lse - rowp[cls];
        bool pos = cls > 0;
        rv = pos ? 0.0f : ce;
        if (pos) {
            np = 1; pce = ce;
            const float* pr = priors + (size_t)p * 4;
            const float* ld = loc + bp * 4;
            float lt0 = (tg[1] - pr[0]) / (pr[2] + 0.1f);
            float lt1 = (tg[2] - pr[1]) / (pr[3] + 0.1f);
            float lt2 = __logf(tg[3] / pr[2]) / 0.2f;
            float lt3 = __logf(tg[4] / pr[3]) / 0.2f;
            float lt[4] = {lt0, lt1, lt2, lt3};
            #pragma unroll
            for (int j = 0; j < 4; ++j) {
                float d = ld[j] - lt[j];
                float ad = fabsf(d);
                ll += (ad < 1.0f) ? 0.5f * d * d : ad - 0.5f;
            }
        }
        rank[bp] = rv;
    }

    // deterministic fixed-order block reduction
    rL[tid] = ll; rC[tid] = pce; rN[tid] = np;
    __syncthreads();
    for (int s = 128; s > 0; s >>= 1) {
        if (tid < s) { rL[tid] += rL[tid + s]; rC[tid] += rC[tid + s]; rN[tid] += rN[tid + s]; }
        __syncthreads();
    }
    if (tid == 0) {
        partialL[b * GY3 + blockIdx.y] = rL[0];
        partialC[b * GY3 + blockIdx.y] = rC[0];
        atomicAdd(&numpos[b], rN[0]);   // integer -> deterministic
    }
}

// ---------------- hard-negative mining: LDS bitonic sort + top-k sum ----------------
__global__ void k_mine(const float* __restrict__ rank,
                       const int* __restrict__ numpos,
                       float* __restrict__ negsum) {
    __shared__ float s[PP];            // 64 KB of the 320 KB WGP LDS
    __shared__ float red[MINE_T];
    const int b = blockIdx.x;
    const int tid = threadIdx.x;

    for (int i = tid; i < PP; i += MINE_T) s[i] = rank[(size_t)b * PP + i];
    __syncthreads();

    for (int k = 2; k <= PP; k <<= 1) {
        for (int j = k >> 1; j > 0; j >>= 1) {
            for (int i = tid; i < PP; i += MINE_T) {
                int ixj = i ^ j;
                if (ixj > i) {
                    float a = s[i], c = s[ixj];
                    bool desc = ((i & k) == 0);
                    if (desc ? (a < c) : (a > c)) { s[i] = c; s[ixj] = a; }
                }
            }
            __syncthreads();
        }
    }

    int np = numpos[b];
    int kk = min(3 * np, PP - 1);
    float sum = 0.0f;
    for (int i = tid; i < kk; i += MINE_T) sum += s[i];
    red[tid] = sum;
    __syncthreads();
    for (int r = MINE_T / 2; r > 0; r >>= 1) {
        if (tid < r) red[tid] += red[tid + r];
        __syncthreads();
    }
    if (tid == 0) negsum[b] = red[0];
}

// ---------------- final serial (deterministic) reduce ----------------
__global__ void k_final(const float* __restrict__ partialL,
                        const float* __restrict__ partialC,
                        const int* __restrict__ numpos,
                        const float* __restrict__ negsum,
                        float* __restrict__ out) {
    float tl = 0.0f, tc = 0.0f, tn = 0.0f; int npos = 0;
    for (int i = 0; i < BB * GY3; ++i) { tl += partialL[i]; tc += partialC[i]; }
    for (int i = 0; i < BB; ++i)       { npos += numpos[i]; tn += negsum[i]; }
    float N = (float)npos;
    out[0] = tl / N;
    out[1] = (tc + tn) / N;
}

extern "C" void kernel_launch(void* const* d_in, const int* in_sizes, int n_in,
                              void* d_out, int out_size, void* d_ws, size_t ws_size,
                              hipStream_t stream) {
    const float* loc     = (const float*)d_in[0];   // [B,P,4]
    const float* conf    = (const float*)d_in[1];   // [B,P,81]
    const float* priors  = (const float*)d_in[2];   // [P,4]
    const float* targets = (const float*)d_in[3];   // [B,O,5]
    float* out = (float*)d_out;

    char* ws = (char*)d_ws;
    const size_t BP = (size_t)BB * PP;
    float* bto       = (float*)ws;                       ws += BP * 4;
    int*   bti       = (int*)ws;                         ws += BP * 4;
    float* rank      = (float*)ws;                       ws += BP * 4;
    u64*   bestPrior = (u64*)ws;                         ws += (size_t)BB * OO * 8;
    float* partialL  = (float*)ws;                       ws += (size_t)BB * GY3 * 4;
    float* partialC  = (float*)ws;                       ws += (size_t)BB * GY3 * 4;
    int*   numpos    = (int*)ws;                         ws += (size_t)BB * 4;
    float* negsum    = (float*)ws;

    k_init <<<1, 512, 0, stream>>>(bestPrior, numpos);
    k_match<<<dim3(BB, PP / 256), 256, 0, stream>>>(priors, targets, bto, bti, bestPrior);
    k_force<<<BB, 1, 0, stream>>>(bestPrior, bto, bti);
    k_loss <<<dim3(BB, GY3), 256, 0, stream>>>(conf, loc, priors, targets,
                                               bto, bti, rank, partialL, partialC, numpos);
    k_mine <<<BB, MINE_T, 0, stream>>>(rank, numpos, negsum);
    k_final<<<1, 1, 0, stream>>>(partialL, partialC, numpos, negsum, out);
}
